// MOELayer_36326833390080
// MI455X (gfx1250) — matrix operand order repaired and checked by reading, compiled, not run on hardware
//
#include <hip/hip_runtime.h>
#include <hip/hip_bf16.h>
#include <math.h>

#define N_TOK 8192
#define DIM   1024
#define HID   4096
#define NEXP  8
#define CAP   2048   // TOP_K * N / E, capacity_factor = 1.0

typedef __attribute__((ext_vector_type(16))) __bf16 v16bf;
typedef __attribute__((ext_vector_type(8)))  float  v8f;

union FragBF { uint4 u[2]; v16bf v; };

__device__ __forceinline__ unsigned short f2bf(float f) {
  unsigned u = __float_as_uint(f);
  u += 0x7fffu + ((u >> 16) & 1u);   // round-to-nearest-even
  return (unsigned short)(u >> 16);
}

// CDNA5 async Global->LDS copy (ASYNCcnt-tracked), per-lane 16B.
__device__ __forceinline__ void async_load_b128(unsigned lds_off,
                                                const void* gptr) {
  asm volatile("global_load_async_to_lds_b128 %0, %1, off"
               :: "v"(lds_off), "v"(gptr) : "memory");
}
__device__ __forceinline__ void wait_async0() {
  asm volatile("s_wait_asynccnt 0x0" ::: "memory");
}

// ---------------------------------------------------------------- init ----
__global__ void init_kernel(unsigned int* __restrict__ disp32) {
  const size_t stride = (size_t)gridDim.x * blockDim.x;
  size_t i = (size_t)blockIdx.x * blockDim.x + threadIdx.x;
  const size_t ndisp = (size_t)NEXP * CAP * DIM / 2;  // bf16 pairs as u32
  for (size_t j = i; j < ndisp; j += stride) disp32[j] = 0u;
}

// -------------------------------------------------------------- gating ----
__global__ __launch_bounds__(256)
void gate_kernel(const float* __restrict__ x, const float* __restrict__ wg,
                 int* __restrict__ top_i, float* __restrict__ gts) {
  const int tid = threadIdx.x, w = tid >> 5, lane = tid & 31;
  const int n = blockIdx.x * 8 + w;
  float acc[NEXP];
#pragma unroll
  for (int e = 0; e < NEXP; ++e) acc[e] = 0.f;
  const float* xr = x + (size_t)n * DIM;
  for (int d = lane; d < DIM; d += 32) {
    const float xv = xr[d];
    const float* wr = wg + d * NEXP;
#pragma unroll
    for (int e = 0; e < NEXP; ++e) acc[e] += xv * wr[e];
  }
#pragma unroll
  for (int off = 16; off >= 1; off >>= 1) {
#pragma unroll
    for (int e = 0; e < NEXP; ++e) acc[e] += __shfl_xor(acc[e], off, 32);
  }
  if (lane == 0) {
    int i1 = 0; float v1 = acc[0];
#pragma unroll
    for (int e = 1; e < NEXP; ++e) if (acc[e] > v1) { v1 = acc[e]; i1 = e; }
    int i2 = -1; float v2 = -3.4e38f;
#pragma unroll
    for (int e = 0; e < NEXP; ++e)
      if (e != i1 && acc[e] > v2) { v2 = acc[e]; i2 = e; }
    // renormalized top-2 softmax gates: p1 = e^l1/(e^l1+e^l2)
    const float p1 = 1.f / (1.f + __expf(v2 - v1));
    top_i[n * 2 + 0] = i1;  top_i[n * 2 + 1] = i2;
    gts[n * 2 + 0] = p1;    gts[n * 2 + 1] = 1.f - p1;
  }
}

// ------------------------------------------- GShard slot-major positions ----
__global__ void pos_kernel(const int* __restrict__ top_i,
                           int* __restrict__ slot_pos) {
  const int lane = threadIdx.x;       // single wave32
  int cnt[NEXP];
#pragma unroll
  for (int e = 0; e < NEXP; ++e) cnt[e] = 0;
  const unsigned below = (1u << lane) - 1u;
  for (int base = 0; base < 2 * N_TOK; base += 32) {
    const int idx = base + lane;
    const int s = idx >> 13;          // idx / 8192
    const int n = idx & (N_TOK - 1);
    const int e = top_i[n * 2 + s];
    int pos = -1;
#pragma unroll
    for (int ee = 0; ee < NEXP; ++ee) {
      const unsigned m = (unsigned)__ballot(e == ee);
      if (e == ee) pos = cnt[ee] + __popc(m & below);
      cnt[ee] += __popc(m);
    }
    slot_pos[n * 2 + s] = (pos < CAP) ? pos : -1;   // -1 == dropped
  }
}

// ------------------------------------------------------------- dispatch ----
__global__ __launch_bounds__(256)
void dispatch_kernel(const float* __restrict__ x, const int* __restrict__ top_i,
                     const int* __restrict__ slot_pos,
                     unsigned short* __restrict__ disp) {
  const int tid = threadIdx.x, w = tid >> 5, lane = tid & 31;
  const int idx = blockIdx.x * 8 + w;          // slot in [0, 2N)
  const int n = idx >> 1, s = idx & 1;
  const int p = slot_pos[n * 2 + s];
  if (p < 0) return;
  const int e = top_i[n * 2 + s];
  const float* xr = x + (size_t)n * DIM;
  unsigned short* dr = disp + ((size_t)e * CAP + p) * DIM;
  for (int d = lane * 4; d < DIM; d += 128) {
    const float4 f = *(const float4*)(xr + d);
    const unsigned u0 = (unsigned)f2bf(f.x) | ((unsigned)f2bf(f.y) << 16);
    const unsigned u1 = (unsigned)f2bf(f.z) | ((unsigned)f2bf(f.w) << 16);
    *(uint2*)(dr + d) = make_uint2(u0, u1);
  }
}

// ------------------------------------------------- WMMA GEMM tile core ----
// Block: 256 threads (8 waves). Output tile 256(M) x 64(N), BK = 32.
// Wave w owns M-rows [32w, 32w+32) as two 16-row M-blocks; 4 N-tiles of 16.
// Per K-step per wave: 4 A ds_load_b128 + 8 B ds_load_b128 -> 8 WMMAs.
// A: bf16 row-major [rows][KA], staged via async global->LDS (ASYNCcnt).
// B: f32 row-major [KA][NB], converted to bf16, stored transposed BT[n][k].
template<int KA, int NB>
__device__ __forceinline__ void gemm_core(const unsigned short* __restrict__ Ae,
                                          const float* __restrict__ Be,
                                          int c0, int n0,
                                          unsigned short* Alds,
                                          unsigned short* BTlds,
                                          v8f acc[2][4]) {
  const int tid  = threadIdx.x;
  const int w    = tid >> 5;
  const int lane = tid & 31;
  const int half = lane >> 4;     // 0: lanes 0-15, 1: lanes 16-31
  const int l15  = lane & 15;

  for (int k0 = 0; k0 < KA; k0 += 32) {
    {   // stage A: 256 rows x 32 bf16 (64 B/row), one row per thread, async
      const unsigned short* src = Ae + (size_t)(c0 + tid) * KA + k0;
      const unsigned lo = (unsigned)(size_t)&Alds[tid * 32];
      async_load_b128(lo,      src);
      async_load_b128(lo + 16, src + 8);
      async_load_b128(lo + 32, src + 16);
      async_load_b128(lo + 48, src + 24);
    }
    {   // stage B: 32 k-rows x 64 cols, f32 -> packed bf16 pairs, BT[n][k]
      const int kp = (tid >> 4) * 2;        // even k
      const int nb = (tid & 15) * 4;        // 4 consecutive columns
      const float* brow0 = Be + (size_t)(k0 + kp) * NB + n0 + nb;
      const float* brow1 = brow0 + NB;
      const float4 f0 = *(const float4*)brow0;
      const float4 f1 = *(const float4*)brow1;
      unsigned* bt = (unsigned*)BTlds;      // u32 view: [n][16]
      const int kw = kp >> 1;
      bt[(nb + 0) * 16 + kw] = (unsigned)f2bf(f0.x) | ((unsigned)f2bf(f1.x) << 16);
      bt[(nb + 1) * 16 + kw] = (unsigned)f2bf(f0.y) | ((unsigned)f2bf(f1.y) << 16);
      bt[(nb + 2) * 16 + kw] = (unsigned)f2bf(f0.z) | ((unsigned)f2bf(f1.z) << 16);
      bt[(nb + 3) * 16 + kw] = (unsigned)f2bf(f0.w) | ((unsigned)f2bf(f1.w) << 16);
      if (k0 + 32 < KA)
        __builtin_prefetch((const void*)(brow0 + (size_t)32 * NB), 0, 0);
    }
    wait_async0();        // ASYNCcnt == 0: A tile resident in LDS
    __syncthreads();

    // A fragment (16x32 bf16): lanes<16: V0-3 K=0..7, V4-7 K=16..23;
    // lanes>=16: K=8..15 / K=24..31. Row m = lane%16.
    FragBF a[2];
#pragma unroll
    for (int mb = 0; mb < 2; ++mb) {
      const int m = 32 * w + 16 * mb + l15;
      a[mb].u[0] = *(const uint4*)&Alds[m * 32 + half * 8];
      a[mb].u[1] = *(const uint4*)&Alds[m * 32 + half * 8 + 16];
    }
#pragma unroll
    for (int th = 0; th < 4; ++th) {
      // B fragment (32x16): lane holds column n = lane%16;
      // lanes<16: K=0..7,16..23, lanes>=16: K=8..15,24..31 (matches A).
      FragBF b;
      const int n = 16 * th + l15;
      b.u[0] = *(const uint4*)&BTlds[n * 32 + half * 8];
      b.u[1] = *(const uint4*)&BTlds[n * 32 + half * 8 + 16];
#pragma unroll
      for (int mb = 0; mb < 2; ++mb)
        acc[mb][th] = __builtin_amdgcn_wmma_f32_16x16x32_bf16(
            false, a[mb].v, false, b.v, (short)0, acc[mb][th], false, false);
    }
    __syncthreads();
  }
}

// ------------------------------------------------- GEMM1: h = relu(..) ----
__global__ __launch_bounds__(256)
void gemm1_kernel(const unsigned short* __restrict__ disp,
                  const float* __restrict__ w1,
                  const float* __restrict__ b1,
                  unsigned short* __restrict__ h) {
  __shared__ unsigned short Alds[256 * 32];
  __shared__ unsigned short BTlds[64 * 32];
  const int e  = blockIdx.z;
  const int c0 = blockIdx.y * 256;
  const int n0 = blockIdx.x * 64;
  v8f acc[2][4];
#pragma unroll
  for (int mb = 0; mb < 2; ++mb)
#pragma unroll
    for (int t = 0; t < 4; ++t)
#pragma unroll
      for (int i = 0; i < 8; ++i) acc[mb][t][i] = 0.f;

  gemm_core<DIM, HID>(disp + (size_t)e * CAP * DIM,
                      w1 + (size_t)e * DIM * HID, c0, n0, Alds, BTlds, acc);

  const int tid = threadIdx.x, w = tid >> 5, lane = tid & 31;
  const int half = lane >> 4, l15 = lane & 15;
  const float* be = b1 + e * HID;
  unsigned short* he = h + (size_t)e * CAP * HID;
#pragma unroll
  for (int th = 0; th < 4; ++th) {
    const int col = n0 + 16 * th + l15;
    const float bias = be[col];
#pragma unroll
    for (int mb = 0; mb < 2; ++mb) {
#pragma unroll
      for (int r = 0; r < 8; ++r) {
        // C/D layout: lanes<16 -> M=r, lanes>=16 -> M=r+8; N = lane%16
        const int c = c0 + 32 * w + 16 * mb + r + half * 8;
        float v = acc[mb][th][r] + bias;
        v = v > 0.f ? v : 0.f;
        he[(size_t)c * HID + col] = f2bf(v);
      }
    }
  }
}

// ------------------------------------------------- GEMM2: y = h*w2 + b2 ----
__global__ __launch_bounds__(256)
void gemm2_kernel(const unsigned short* __restrict__ h,
                  const float* __restrict__ w2,
                  const float* __restrict__ b2,
                  float* __restrict__ y) {
  __shared__ unsigned short Alds[256 * 32];
  __shared__ unsigned short BTlds[64 * 32];
  const int e  = blockIdx.z;
  const int c0 = blockIdx.y * 256;
  const int n0 = blockIdx.x * 64;
  v8f acc[2][4];
#pragma unroll
  for (int mb = 0; mb < 2; ++mb)
#pragma unroll
    for (int t = 0; t < 4; ++t)
#pragma unroll
      for (int i = 0; i < 8; ++i) acc[mb][t][i] = 0.f;

  gemm_core<HID, DIM>(h + (size_t)e * CAP * HID,
                      w2 + (size_t)e * HID * DIM, c0, n0, Alds, BTlds, acc);

  const int tid = threadIdx.x, w = tid >> 5, lane = tid & 31;
  const int half = lane >> 4, l15 = lane & 15;
  const float* be = b2 + e * DIM;
  float* ye = y + (size_t)e * CAP * DIM;
#pragma unroll
  for (int th = 0; th < 4; ++th) {
    const int col = n0 + 16 * th + l15;
    const float bias = be[col];
#pragma unroll
    for (int mb = 0; mb < 2; ++mb) {
#pragma unroll
      for (int r = 0; r < 8; ++r) {
        const int c = c0 + 32 * w + 16 * mb + r + half * 8;
        ye[(size_t)c * DIM + col] = acc[mb][th][r] + bias;
      }
    }
  }
}

// -------------------------------------------------------------- combine ----
__global__ __launch_bounds__(256)
void combine_kernel(const float* __restrict__ y, const int* __restrict__ top_i,
                    const int* __restrict__ slot_pos,
                    const float* __restrict__ gts, float* __restrict__ out) {
  const int tid = threadIdx.x, w = tid >> 5, lane = tid & 31;
  const int n = blockIdx.x * 8 + w;
  const int p0 = slot_pos[n * 2 + 0], p1 = slot_pos[n * 2 + 1];
  const int e0 = top_i[n * 2 + 0],   e1 = top_i[n * 2 + 1];
  const float g0 = (p0 >= 0) ? gts[n * 2 + 0] : 0.f;
  const float g1 = (p1 >= 0) ? gts[n * 2 + 1] : 0.f;
  const float* y0 = y + ((p0 >= 0) ? ((size_t)e0 * CAP + p0) * DIM : 0);
  const float* y1 = y + ((p1 >= 0) ? ((size_t)e1 * CAP + p1) * DIM : 0);
  float* orow = out + (size_t)n * DIM;
  for (int d = lane * 4; d < DIM; d += 128) {
    const float4 a = *(const float4*)(y0 + d);
    const float4 b = *(const float4*)(y1 + d);
    float4 o;
    o.x = g0 * a.x + g1 * b.x;
    o.y = g0 * a.y + g1 * b.y;
    o.z = g0 * a.z + g1 * b.z;
    o.w = g0 * a.w + g1 * b.w;
    *(float4*)(orow + d) = o;
  }
}

// ---------------------------------------------------------------- launch ----
extern "C" void kernel_launch(void* const* d_in, const int* in_sizes, int n_in,
                              void* d_out, int out_size, void* d_ws,
                              size_t ws_size, hipStream_t stream) {
  (void)in_sizes; (void)n_in; (void)out_size; (void)ws_size;
  const float* x  = (const float*)d_in[0];
  const float* wg = (const float*)d_in[1];
  const float* w1 = (const float*)d_in[2];
  const float* b1 = (const float*)d_in[3];
  const float* w2 = (const float*)d_in[4];
  const float* b2 = (const float*)d_in[5];
  float* out = (float*)d_out;

  char* ws = (char*)d_ws;
  size_t off = 0;
  auto alloc = [&](size_t bytes) -> void* {
    void* p = ws + off;
    off = (off + bytes + 255) & ~(size_t)255;
    return p;
  };
  int*   top_i    = (int*)  alloc((size_t)N_TOK * 2 * sizeof(int));
  float* gts      = (float*)alloc((size_t)N_TOK * 2 * sizeof(float));
  int*   slot_pos = (int*)  alloc((size_t)N_TOK * 2 * sizeof(int));
  unsigned short* disp = (unsigned short*)alloc((size_t)NEXP * CAP * DIM * 2);
  unsigned short* hbuf = (unsigned short*)alloc((size_t)NEXP * CAP * HID * 2);
  float*          ybuf = (float*)alloc((size_t)NEXP * CAP * DIM * sizeof(float));

  init_kernel<<<dim3(2048), dim3(256), 0, stream>>>((unsigned int*)disp);
  gate_kernel<<<dim3(N_TOK / 8), dim3(256), 0, stream>>>(x, wg, top_i, gts);
  pos_kernel<<<dim3(1), dim3(32), 0, stream>>>(top_i, slot_pos);
  dispatch_kernel<<<dim3(2 * N_TOK / 8), dim3(256), 0, stream>>>(x, top_i,
                                                                 slot_pos, disp);
  gemm1_kernel<<<dim3(HID / 64, CAP / 256, NEXP), dim3(256), 0, stream>>>(
      disp, w1, b1, hbuf);
  gemm2_kernel<<<dim3(DIM / 64, CAP / 256, NEXP), dim3(256), 0, stream>>>(
      hbuf, w2, b2, ybuf);
  combine_kernel<<<dim3(N_TOK / 8), dim3(256), 0, stream>>>(ybuf, top_i,
                                                            slot_pos, gts, out);
}
